// SemanticIDQuantizer_18090402251369
// MI455X (gfx1250) — matrix operand order repaired and checked by reading, not compiled, not run on hardware
//
#include <hip/hip_runtime.h>
#include <hip/hip_bf16.h>

// Problem constants
#define BDIM 8192
#define HID  512
#define KCB  4096
#define LQ   4
#define WPB  8            // waves per block in the layer kernel
#define CT_PER_WAVE (256 / WPB)

typedef __attribute__((ext_vector_type(16))) __bf16 v16bf;
typedef __attribute__((ext_vector_type(8)))  __bf16 v8bf;
typedef __attribute__((ext_vector_type(8)))  float  v8f;

// Load 16 bf16 elements for a WMMA operand as two contiguous 16B chunks
// (chunk1 at p, chunk2 at p+stride2 elements), packed in vector-element order.
__device__ __forceinline__ v16bf ld_op16(const __bf16* p, int stride2) {
  v8bf lo = *(const v8bf*)(p);
  v8bf hi = *(const v8bf*)(p + stride2);
  return __builtin_shufflevector(lo, hi, 0,1,2,3,4,5,6,7,8,9,10,11,12,13,14,15);
}

// ---------------------------------------------------------------------------
// Prep: x -> bf16 residual buffer; zero codebook_usage output region
// ---------------------------------------------------------------------------
__global__ void prep_kernel(const float* __restrict__ x, __bf16* __restrict__ resb,
                            float* __restrict__ usage, int nconv, int nusage) {
  int i = blockIdx.x * blockDim.x + threadIdx.x;
  int stride = gridDim.x * blockDim.x;
  for (int j = i; j < nconv; j += stride) resb[j] = (__bf16)x[j];
  for (int j = i; j < nusage; j += stride) usage[j] = 0.0f;
}

// ---------------------------------------------------------------------------
// Convert one layer's fp32 codebook to bf16 (L2-resident operand for WMMA)
// ---------------------------------------------------------------------------
__global__ void cvt_cb_kernel(const float* __restrict__ cb, __bf16* __restrict__ cbb, int n) {
  int i = blockIdx.x * blockDim.x + threadIdx.x;
  int stride = gridDim.x * blockDim.x;
  for (int j = i; j < n; j += stride) cbb[j] = (__bf16)cb[j];
}

// ---------------------------------------------------------------------------
// Per-layer: WMMA similarity + row argmax + exact fp32 residual update.
// Block = 8 waves, all covering the same two 16-row strips (rows 32*blk..+31).
// Each wave holds both A strips in VGPRs (2*128) and sweeps 32 of the 256
// column tiles; per-wave argmax candidates are merged in LDS.
// ---------------------------------------------------------------------------
__global__ __launch_bounds__(WPB * 32) void rvq_layer_kernel(
    const float* __restrict__ x,         // [B, D] original input
    const float* __restrict__ cbf,       // [K, D] fp32 codebook, this layer
    const __bf16* __restrict__ cbb,      // [K, D] bf16 codebook, this layer
    __bf16* __restrict__ resb,           // [B, D] bf16 residual (in: this layer, out: next)
    float* __restrict__ quant,           // [B, D] output quantized (accumulated)
    float* __restrict__ resStack,        // [B, L, D] output residuals
    float* __restrict__ usage,           // [L, K] output usage
    float* __restrict__ lossP,           // [B/32] per-block loss partials
    int layer) {
  const int tid     = threadIdx.x;
  const int lane    = tid & 31;
  // Wave index as a proven-uniform (SGPR) value: keeps the tile loop scalar,
  // avoiding divergent-loop exec bookkeeping around the WMMA body.
  const int wave    = __builtin_amdgcn_readfirstlane(tid >> 5);
  const int m16     = lane & 15;
  const int hi      = (lane >> 4) & 1;          // 0: lanes 0-15, 1: lanes 16-31
  const int rowBase = blockIdx.x * 32;
  const int rowA    = rowBase + m16;            // strip A source row for this lane
  const int rowB    = rowA + 16;                // strip B source row
  const int koffA   = hi ? 8 : 0;               // A: K interleave {0..7,16..23}/{8..15,24..31}
  const int koffB   = hi ? 8 : 0;               // B: K interleave {0..7,16..23}/{8..15,24..31}

  __shared__ float smax[WPB * 32];
  __shared__ int   sidxw[WPB * 32];
  __shared__ int   sfinal[32];
  __shared__ float sred[WPB * 32];

  // ---- Load both A strips (16x512 bf16 each) fully into VGPRs: 2*128 VGPRs ----
  v16bf A0[16], A1[16];
#pragma unroll
  for (int dc = 0; dc < 16; ++dc) {
    A0[dc] = ld_op16(resb + (size_t)rowA * HID + dc * 32 + koffA, 16);
    A1[dc] = ld_op16(resb + (size_t)rowB * HID + dc * 32 + koffA, 16);
  }

  float m0[8], m1[8];
  int   i0[8], i1[8];
#pragma unroll
  for (int e = 0; e < 8; ++e) {
    m0[e] = -__builtin_inff(); m1[e] = -__builtin_inff();
    i0[e] = 0;                 i1[e] = 0;
  }

  // ---- Each wave sweeps its 32 column tiles; 32 WMMAs per tile ----
  const int ctBeg = wave * CT_PER_WAVE;
  const int ctEnd = ctBeg + CT_PER_WAVE;
  const __bf16* bptr = cbb + (size_t)(ctBeg * 16 + m16) * HID + koffB;
  for (int ct = ctBeg; ct < ctEnd; ++ct, bptr += 16 * HID) {
    v8f c0 = {}; v8f c1 = {};
#pragma unroll
    for (int dc = 0; dc < 16; ++dc) {
      v16bf b = ld_op16(bptr + dc * 32, 16);
      c0 = __builtin_amdgcn_wmma_f32_16x16x32_bf16(false, A0[dc], false, b, (short)0, c0, false, false);
      c1 = __builtin_amdgcn_wmma_f32_16x16x32_bf16(false, A1[dc], false, b, (short)0, c1, false, false);
    }
    const int cand = ct * 16 + m16;   // C-matrix column for this lane
#pragma unroll
    for (int e = 0; e < 8; ++e) {
      if (c0[e] > m0[e]) { m0[e] = c0[e]; i0[e] = cand; }
      if (c1[e] > m1[e]) { m1[e] = c1[e]; i1[e] = cand; }
    }
  }

  // ---- Per-wave cross-lane argmax within each 16-lane group ----
#pragma unroll
  for (int e = 0; e < 8; ++e) {
    float m = m0[e]; int id = i0[e];
#pragma unroll
    for (int mk = 8; mk >= 1; mk >>= 1) {
      float om = __shfl_xor(m, mk, 32);
      int   oi = __shfl_xor(id, mk, 32);
      if (om > m || (om == m && oi < id)) { m = om; id = oi; }
    }
    if (m16 == 0) { smax[wave * 32 + hi * 8 + e] = m; sidxw[wave * 32 + hi * 8 + e] = id; }

    float n = m1[e]; int jd = i1[e];
#pragma unroll
    for (int mk = 8; mk >= 1; mk >>= 1) {
      float on = __shfl_xor(n, mk, 32);
      int   oj = __shfl_xor(jd, mk, 32);
      if (on > n || (on == n && oj < jd)) { n = on; jd = oj; }
    }
    if (m16 == 0) { smax[wave * 32 + 16 + hi * 8 + e] = n; sidxw[wave * 32 + 16 + hi * 8 + e] = jd; }
  }
  __syncthreads();

  // ---- Merge the WPB per-wave candidates for each of the 32 rows ----
  // Wave w covers ascending column ranges; strictly-greater comparison with
  // smaller-index tie-break preserves jnp.argmax first-max semantics.
  if (tid < 32) {
    float bm = smax[tid];
    int   bi = sidxw[tid];
#pragma unroll
    for (int w = 1; w < WPB; ++w) {
      float om = smax[w * 32 + tid];
      int   oi = sidxw[w * 32 + tid];
      if (om > bm || (om == bm && oi < bi)) { bm = om; bi = oi; }
    }
    sfinal[tid] = bi;
  }
  __syncthreads();

  // ---- Exact fp32 update for the 32 rows owned by this block (256 threads) ----
  float lsum = 0.0f;
  for (int r = 0; r < 32; ++r) {
    const int idx = sfinal[r];
    const size_t g = (size_t)(rowBase + r);
    const float* crow = cbf + (size_t)idx * HID;
    const float* rold = (layer == 0) ? (x + g * HID)
                                     : (resStack + (g * LQ + (layer - 1)) * HID);
    float*  rnewp = resStack + (g * LQ + layer) * HID;
    float*  qrow  = quant + g * HID;
    __bf16* rbrow = resb + g * HID;
#pragma unroll
    for (int j = 0; j < 2; ++j) {
      int d = j * 256 + tid;
      float c  = crow[d];
      float rn = rold[d] - c;
      rnewp[d] = rn;
      if (layer == 0) qrow[d] = c; else qrow[d] += c;
      if (layer != LQ - 1) rbrow[d] = (__bf16)rn;
      lsum += rn * rn;
    }
  }
  if (tid < 32) usage[(size_t)layer * KCB + sfinal[tid]] = 1.0f;  // benign same-value races

  // ---- Deterministic block reduction of the loss partial ----
  sred[tid] = lsum;
  __syncthreads();
  for (int off = (WPB * 32) / 2; off > 0; off >>= 1) {
    if (tid < off) sred[tid] += sred[tid + off];
    __syncthreads();
  }
  if (tid == 0) {
    if (layer == 0) lossP[blockIdx.x] = sred[0];
    else            lossP[blockIdx.x] += sred[0];
  }
}

// ---------------------------------------------------------------------------
// Codebook L2 regularizer partials: sum of squares over all L*K*D elements
// ---------------------------------------------------------------------------
__global__ void reg_kernel(const float* __restrict__ cbs, float* __restrict__ regP, int n) {
  __shared__ float sred[256];
  float s = 0.0f;
  for (int j = blockIdx.x * blockDim.x + threadIdx.x; j < n; j += gridDim.x * blockDim.x) {
    float v = cbs[j];
    s += v * v;
  }
  sred[threadIdx.x] = s;
  __syncthreads();
  for (int off = 128; off > 0; off >>= 1) {
    if (threadIdx.x < off) sred[threadIdx.x] += sred[threadIdx.x + off];
    __syncthreads();
  }
  if (threadIdx.x == 0) regP[blockIdx.x] = sred[0];
}

// ---------------------------------------------------------------------------
// Final scalar: loss = (sum_sq/(B*D) + 0.01 * sum_cb_sq/(L*K)) / D
// ---------------------------------------------------------------------------
__global__ void final_kernel(const float* __restrict__ lossP, const float* __restrict__ regP,
                             float* __restrict__ outLoss) {
  __shared__ float s1[256];
  __shared__ float s2[256];
  int t = threadIdx.x;
  s1[t] = lossP[t];
  s2[t] = regP[t];
  __syncthreads();
  for (int off = 128; off > 0; off >>= 1) {
    if (t < off) { s1[t] += s1[t + off]; s2[t] += s2[t + off]; }
    __syncthreads();
  }
  if (t == 0) {
    float loss = s1[0] / ((float)BDIM * (float)HID)
               + 0.01f * (s2[0] / ((float)LQ * (float)KCB));
    outLoss[0] = loss / (float)HID;
  }
}

// ---------------------------------------------------------------------------
extern "C" void kernel_launch(void* const* d_in, const int* in_sizes, int n_in,
                              void* d_out, int out_size, void* d_ws, size_t ws_size,
                              hipStream_t stream) {
  const float* x   = (const float*)d_in[0];
  // d_in[1] = temperature: scale-invariant under argmax; no forward-value effect.
  const float* cbs = (const float*)d_in[2];

  float* out      = (float*)d_out;
  float* quant    = out;                                   // [B, D]
  float* outLoss  = out + (size_t)BDIM * HID;              // scalar
  float* resStack = outLoss + 1;                           // [B, L, D]
  float* usage    = resStack + (size_t)BDIM * LQ * HID;    // [L, K]

  char* ws = (char*)d_ws;
  __bf16* resb  = (__bf16*)ws;                                       // 8 MB
  __bf16* cbb   = (__bf16*)(ws + (size_t)BDIM * HID * 2);            // 4 MB
  float*  lossP = (float*)(ws + (size_t)BDIM * HID * 2 + (size_t)KCB * HID * 2);
  float*  regP  = lossP + 256;

  prep_kernel<<<512, 256, 0, stream>>>(x, resb, usage, BDIM * HID, LQ * KCB);
  for (int i = 0; i < LQ; ++i) {
    const float* cbf = cbs + (size_t)i * KCB * HID;
    cvt_cb_kernel<<<512, 256, 0, stream>>>(cbf, cbb, KCB * HID);
    rvq_layer_kernel<<<BDIM / 32, WPB * 32, 0, stream>>>(x, cbf, cbb, resb, quant,
                                                         resStack, usage, lossP, i);
  }
  reg_kernel<<<256, 256, 0, stream>>>(cbs, regP, LQ * KCB * HID);
  final_kernel<<<1, 256, 0, stream>>>(lossP, regP, outLoss);
}
